// MultiHeadSelfAttention2D_39728447488110
// MI455X (gfx1250) — matrix operand order repaired and checked by reading, not compiled, not run on hardware
//
#include <hip/hip_runtime.h>

#define DIM      512
#define HEADS    8
#define HEAD_DIM 64
#define NTOK     2304   // 48*48
#define BATCH    4
#define SCALE    0.125f // 64^-0.5

typedef __attribute__((ext_vector_type(16))) __bf16 v16bf;
typedef __attribute__((ext_vector_type(8)))  __bf16 v8bf;
typedef __attribute__((ext_vector_type(8)))  float  v8f;

// D = A(16x32 bf16) * B(32x16 bf16) + C(16x16 f32)
static __device__ __forceinline__ v8f wmma_bf16(v16bf a, v16bf b, v8f c) {
    return __builtin_amdgcn_wmma_f32_16x16x32_bf16(
        /*neg_a=*/false, a, /*neg_b=*/false, b,
        /*c_mod=*/(short)0, c, /*reuse_a=*/false, /*reuse_b=*/false);
}

// A-fragment (16x32, 16-bit): lane holds row (lane%16); elements 0..15 are
// K = lh*16..lh*16+15 (contiguous order, matching the B fragments).
// p points at (row start + k_base). Two contiguous 16B loads per lane.
static __device__ __forceinline__ v16bf load_afrag(const __bf16* p, int lh) {
    v8bf lo = *(const v8bf*)(p + lh * 16);
    v8bf hi = *(const v8bf*)(p + lh * 16 + 8);
    return __builtin_shufflevector(lo, hi, 0,1,2,3,4,5,6,7,8,9,10,11,12,13,14,15);
}

// ---------------------------------------------------------------------------
// Kernel 0: weights f32 -> bf16
__global__ void cvt_weights(const float* __restrict__ qkv_w,
                            const float* __restrict__ proj_w,
                            __bf16* __restrict__ wq, __bf16* __restrict__ wp) {
    int i = blockIdx.x * 256 + threadIdx.x;
    const int NQ = 3 * DIM * DIM;
    if (i < NQ) wq[i] = (__bf16)qkv_w[i];
    else        wp[i - NQ] = (__bf16)proj_w[i - NQ];
}

// ---------------------------------------------------------------------------
// Kernel 1: x [b][c][n] f32 -> xT [b][n][c] bf16 (LDS tile transpose)
__global__ void xpose(const float* __restrict__ x, __bf16* __restrict__ xT) {
    __shared__ float tile[32][33];
    int b = blockIdx.z;
    int c0 = blockIdx.y * 32, n0 = blockIdx.x * 32;
    int tx = threadIdx.x, ty = threadIdx.y;           // block (32, 8)
    const float* xb = x + (size_t)b * DIM * NTOK;
#pragma unroll
    for (int k = 0; k < 4; ++k)
        tile[ty + 8 * k][tx] = xb[(size_t)(c0 + ty + 8 * k) * NTOK + n0 + tx];
    __syncthreads();
    __bf16* xTb = xT + (size_t)b * NTOK * DIM;
#pragma unroll
    for (int k = 0; k < 4; ++k)
        xTb[(size_t)(n0 + ty + 8 * k) * DIM + c0 + tx] = (__bf16)tile[tx][ty + 8 * k];
}

// ---------------------------------------------------------------------------
// Kernel 2: QKV GEMM. qkv[o,n] = sum_c wq[o,c] * xT[n,c].
// Wave computes 32(M=o) x 64(N=n): 2 A-frags x 4 B-frags = 8 WMMA / k-step.
// Two-buffer ping-pong, fully unrolled: each half-step consumes a buffer and
// immediately reissues loads into that same buffer for k+64 -> no reg moves,
// one-chunk load lookahead at all times.
__global__ __launch_bounds__(128) void qkv_gemm(
        const __bf16* __restrict__ wq, const __bf16* __restrict__ xT,
        __bf16* __restrict__ Qb, __bf16* __restrict__ Kb, __bf16* __restrict__ Vt) {
    int lane = threadIdx.x & 31, wave = threadIdx.x >> 5;
    int lr = lane & 15, lh = lane >> 4;
    int b  = blockIdx.z;
    int o0 = blockIdx.y * 128 + wave * 32;
    int n0 = blockIdx.x * 64;

    const __bf16* arow0 = wq + (size_t)(o0 + lr) * DIM;
    const __bf16* arow1 = wq + (size_t)(o0 + 16 + lr) * DIM;
    const __bf16* xrow  = xT + ((size_t)b * NTOK + n0 + lr) * DIM;

    v8f acc[2][4] = {};
    auto loadk = [&](int k, v16bf a[2], v16bf bb[4]) {
        a[0] = load_afrag(arow0 + k, lh);
        a[1] = load_afrag(arow1 + k, lh);
#pragma unroll
        for (int cf = 0; cf < 4; ++cf)
            bb[cf] = *(const v16bf*)(xrow + (size_t)cf * 16 * DIM + k + lh * 16);
    };
    auto mmas = [&](v16bf a[2], v16bf bb[4]) {
#pragma unroll
        for (int mi = 0; mi < 2; ++mi)
#pragma unroll
            for (int cf = 0; cf < 4; ++cf)
                acc[mi][cf] = wmma_bf16(a[mi], bb[cf], acc[mi][cf]);
    };

    v16bf a0[2], b0[4], a1[2], b1[4];
    loadk(0,  a0, b0);
    loadk(32, a1, b1);
#pragma unroll
    for (int k = 0; k < DIM - 64; k += 64) {
        mmas(a0, b0);
        loadk(k + 64, a0, b0);
        mmas(a1, b1);
        loadk(k + 96, a1, b1);
    }
    mmas(a0, b0);
    mmas(a1, b1);

    int seg  = o0 >> 9;                 // 0:Q 1:K 2:V (32-row tile never spans)
    int oo   = o0 & 511;
    int head = oo >> 6;                 // 32-row tile stays within one head
    size_t bh = (size_t)b * HEADS + head;
#pragma unroll
    for (int mi = 0; mi < 2; ++mi) {
        int dbase = (oo & 63) + mi * 16 + 8 * lh;   // C rows M=r+8*lh -> contig d
        if (seg < 2) {
            __bf16* dst = (seg == 0 ? Qb : Kb) + bh * NTOK * HEAD_DIM;
#pragma unroll
            for (int cf = 0; cf < 4; ++cf) {
                int n = n0 + cf * 16 + lr;
                v8bf pk;
#pragma unroll
                for (int r = 0; r < 8; ++r) pk[r] = (__bf16)acc[mi][cf][r];
                *(v8bf*)(dst + (size_t)n * HEAD_DIM + dbase) = pk;     // packed 16B
            }
        } else {
            __bf16* dst = Vt + bh * HEAD_DIM * NTOK;                   // [d][n]
#pragma unroll
            for (int cf = 0; cf < 4; ++cf) {
                int n = n0 + cf * 16 + lr;
#pragma unroll
                for (int r = 0; r < 8; ++r)
                    dst[(size_t)(dbase + r) * NTOK + n] = (__bf16)acc[mi][cf][r];
            }
        }
    }
}

// ---------------------------------------------------------------------------
// Kernel 3: flash attention. Wave owns 16 queries; 64-key chunks; online
// softmax with shfl reductions over 16-lane halves; P transposed via LDS.
// V fragments are loaded right after the S WMMAs so their latency hides
// under the softmax VALU work; next chunk is prefetched (global_prefetch_b8).
__global__ __launch_bounds__(128) void attn(
        const __bf16* __restrict__ Qb, const __bf16* __restrict__ Kb,
        const __bf16* __restrict__ Vt, __bf16* __restrict__ Ob) {
    __shared__ __align__(32) __bf16 pbuf[4][16 * 64];   // per-wave P tile
    int lane = threadIdx.x & 31, wave = threadIdx.x >> 5;
    int lr = lane & 15, lh = lane >> 4;
    int b = blockIdx.z, h = blockIdx.y;
    int q0 = blockIdx.x * 64 + wave * 16;
    size_t bh = (size_t)b * HEADS + h;
    const __bf16* qbase = Qb + bh * NTOK * HEAD_DIM;
    const __bf16* kbase = Kb + bh * NTOK * HEAD_DIM;
    const __bf16* vbase = Vt + bh * HEAD_DIM * NTOK;

    v16bf qa[2];
#pragma unroll
    for (int kk = 0; kk < 2; ++kk)
        qa[kk] = load_afrag(qbase + (size_t)(q0 + lr) * HEAD_DIM + kk * 32, lh);

    v8f oacc[4] = {};
    v8f mrow, lrow = {};
#pragma unroll
    for (int r = 0; r < 8; ++r) mrow[r] = -1e30f;
    __bf16* myp = &pbuf[wave][0];

    for (int mc = 0; mc < NTOK; mc += 64) {
        // warm near cache for the next chunk (global_prefetch_b8)
        if (mc + 64 < NTOK) {
            __builtin_prefetch(kbase + (size_t)(mc + 64 + lane) * HEAD_DIM, 0, 3);
            __builtin_prefetch(vbase + (size_t)(lane * 2) * NTOK + mc + 64, 0, 3);
        }
        // S = Q * K^T (scaled)
        v8f s[4];
#pragma unroll
        for (int cf = 0; cf < 4; ++cf) {
            const __bf16* krow = kbase + (size_t)(mc + cf * 16 + lr) * HEAD_DIM;
            v16bf b0 = *(const v16bf*)(krow + lh * 16);        // d = lh*16+e
            v16bf b1 = *(const v16bf*)(krow + 32 + lh * 16);   // d = 32+lh*16+e
            v8f a = {};
            a = wmma_bf16(qa[0], b0, a);
            a = wmma_bf16(qa[1], b1, a);
#pragma unroll
            for (int r = 0; r < 8; ++r) s[cf][r] = a[r] * SCALE;
        }
        // issue V fragment loads now; consumed after the LDS round trip,
        // so their latency hides under the softmax below.
        v16bf vb[4][2];
#pragma unroll
        for (int df = 0; df < 4; ++df) {
            const __bf16* vrow = vbase + (size_t)(df * 16 + lr) * NTOK + mc;
            vb[df][0] = *(const v16bf*)(vrow + lh * 16);
            vb[df][1] = *(const v16bf*)(vrow + 32 + lh * 16);
        }
        // row max: rows live as element r across the 16 lanes of each half
        v8f mnew = mrow;
#pragma unroll
        for (int cf = 0; cf < 4; ++cf)
#pragma unroll
            for (int r = 0; r < 8; ++r) mnew[r] = fmaxf(mnew[r], s[cf][r]);
#pragma unroll
        for (int r = 0; r < 8; ++r) {
            float v = mnew[r];
            v = fmaxf(v, __shfl_xor(v, 1));
            v = fmaxf(v, __shfl_xor(v, 2));
            v = fmaxf(v, __shfl_xor(v, 4));
            v = fmaxf(v, __shfl_xor(v, 8));
            mnew[r] = v;
        }
#pragma unroll
        for (int r = 0; r < 8; ++r) {
            float alpha = __expf(mrow[r] - mnew[r]);
            lrow[r] *= alpha;
#pragma unroll
            for (int df = 0; df < 4; ++df) oacc[df][r] *= alpha;
            mrow[r] = mnew[r];
        }
        // P = exp(S - m), row-sum, transpose C-layout -> LDS row-major tile
        v8f psum = {};
#pragma unroll
        for (int cf = 0; cf < 4; ++cf)
#pragma unroll
            for (int r = 0; r < 8; ++r) {
                float p = __expf(s[cf][r] - mnew[r]);
                psum[r] += p;
                myp[(r + 8 * lh) * 64 + cf * 16 + lr] = (__bf16)p;
            }
#pragma unroll
        for (int r = 0; r < 8; ++r) {
            float v = psum[r];
            v += __shfl_xor(v, 1);
            v += __shfl_xor(v, 2);
            v += __shfl_xor(v, 4);
            v += __shfl_xor(v, 8);
            lrow[r] += v;
        }
        __syncthreads();
        v16bf pa0 = load_afrag(myp + lr * 64, lh);
        v16bf pa1 = load_afrag(myp + lr * 64 + 32, lh);
        // O += P * V  (B from Vt[d][m]: contiguous in m per lane)
#pragma unroll
        for (int df = 0; df < 4; ++df) {
            oacc[df] = wmma_bf16(pa0, vb[df][0], oacc[df]);
            oacc[df] = wmma_bf16(pa1, vb[df][1], oacc[df]);
        }
        __syncthreads();
    }
    // write ObT [b][n][c=h*64+d] bf16
    __bf16* obase = Ob + (size_t)b * NTOK * DIM + (size_t)h * HEAD_DIM;
#pragma unroll
    for (int df = 0; df < 4; ++df) {
        int d = df * 16 + lr;
#pragma unroll
        for (int r = 0; r < 8; ++r) {
            int n = q0 + 8 * lh + r;
            obase[(size_t)n * DIM + d] = (__bf16)(oacc[df][r] / lrow[r]);
        }
    }
}

// ---------------------------------------------------------------------------
// Kernel 4: projection GEMM. out[b][co][n] = sum_c wp[co,c] * ObT[b][n][c]
// Same ping-pong 32x64 wave tile as qkv_gemm; f32 output [b][co][n].
__global__ __launch_bounds__(128) void proj_gemm(
        const __bf16* __restrict__ wp, const __bf16* __restrict__ Ob,
        float* __restrict__ out) {
    int lane = threadIdx.x & 31, wave = threadIdx.x >> 5;
    int lr = lane & 15, lh = lane >> 4;
    int b  = blockIdx.z;
    int o0 = blockIdx.y * 128 + wave * 32;
    int n0 = blockIdx.x * 64;
    const __bf16* arow0 = wp + (size_t)(o0 + lr) * DIM;
    const __bf16* arow1 = wp + (size_t)(o0 + 16 + lr) * DIM;
    const __bf16* brow  = Ob + ((size_t)b * NTOK + n0 + lr) * DIM;

    v8f acc[2][4] = {};
    auto loadk = [&](int k, v16bf a[2], v16bf bb[4]) {
        a[0] = load_afrag(arow0 + k, lh);
        a[1] = load_afrag(arow1 + k, lh);
#pragma unroll
        for (int cf = 0; cf < 4; ++cf)
            bb[cf] = *(const v16bf*)(brow + (size_t)cf * 16 * DIM + k + lh * 16);
    };
    auto mmas = [&](v16bf a[2], v16bf bb[4]) {
#pragma unroll
        for (int mi = 0; mi < 2; ++mi)
#pragma unroll
            for (int cf = 0; cf < 4; ++cf)
                acc[mi][cf] = wmma_bf16(a[mi], bb[cf], acc[mi][cf]);
    };

    v16bf a0[2], b0[4], a1[2], b1[4];
    loadk(0,  a0, b0);
    loadk(32, a1, b1);
#pragma unroll
    for (int k = 0; k < DIM - 64; k += 64) {
        mmas(a0, b0);
        loadk(k + 64, a0, b0);
        mmas(a1, b1);
        loadk(k + 96, a1, b1);
    }
    mmas(a0, b0);
    mmas(a1, b1);

    float* obat = out + (size_t)b * DIM * NTOK;
#pragma unroll
    for (int mi = 0; mi < 2; ++mi)
#pragma unroll
        for (int cf = 0; cf < 4; ++cf) {
            int n = n0 + cf * 16 + lr;
#pragma unroll
            for (int r = 0; r < 8; ++r)
                obat[(size_t)(o0 + mi * 16 + 8 * lh + r) * NTOK + n] = acc[mi][cf][r];
        }
}

// ---------------------------------------------------------------------------
extern "C" void kernel_launch(void* const* d_in, const int* in_sizes, int n_in,
                              void* d_out, int out_size, void* d_ws, size_t ws_size,
                              hipStream_t stream) {
    (void)in_sizes; (void)n_in; (void)out_size; (void)ws_size;
    const float* x      = (const float*)d_in[0];
    const float* qkv_w  = (const float*)d_in[1];
    const float* proj_w = (const float*)d_in[2];
    float* out = (float*)d_out;

    char* ws = (char*)d_ws;
    __bf16* wq = (__bf16*)(ws + 0);          // 1536*512*2  = 1,572,864
    __bf16* wp = (__bf16*)(ws + 1572864);    //  512*512*2  =   524,288
    __bf16* xT = (__bf16*)(ws + 2097152);    // 4*2304*512*2 = 9,437,184
    __bf16* Qb = (__bf16*)(ws + 11534336);   // 9,437,184
    __bf16* Kb = (__bf16*)(ws + 20971520);   // 9,437,184
    __bf16* Vt = (__bf16*)(ws + 30408704);   // 9,437,184
    __bf16* Ob = (__bf16*)(ws + 39845888);   // 9,437,184 -> end 49,283,072

    cvt_weights<<<4096, 256, 0, stream>>>(qkv_w, proj_w, wq, wp);
    xpose<<<dim3(NTOK / 32, DIM / 32, BATCH), dim3(32, 8), 0, stream>>>(x, xT);
    qkv_gemm<<<dim3(NTOK / 64, (3 * DIM) / 128, BATCH), 128, 0, stream>>>(wq, xT, Qb, Kb, Vt);
    attn<<<dim3(NTOK / 64, HEADS, BATCH), 128, 0, stream>>>(Qb, Kb, Vt, Ob);
    proj_gemm<<<dim3(NTOK / 64, DIM / 128, BATCH), 128, 0, stream>>>(wp, Ob, out);
}
